// Attention_90898687853214
// MI455X (gfx1250) — matrix operand order repaired and checked by reading, not compiled, not run on hardware
//
#include <hip/hip_runtime.h>
#include <hip/hip_bf16.h>

typedef __bf16 bf16;
typedef __attribute__((ext_vector_type(16))) __bf16 bf16x16;
typedef __attribute__((ext_vector_type(8)))  __bf16 bf16x8;
typedef __attribute__((ext_vector_type(8)))  float  f32x8;

#define DIMSZ 1024
#define NSEQ  2048
#define BATCH 4
#define NHEAD 16
#define DHEAD 64
#define MROWS (BATCH*NSEQ)     /* 8192 */
#define QKVN  (3*DIMSZ)        /* 3072 */

__device__ __forceinline__ bf16 f2bf(float f) {
  __hip_bfloat16 h = __float2bfloat16(f);
  return __builtin_bit_cast(bf16, h);
}

__device__ __forceinline__ bf16x16 join8(bf16x8 a, bf16x8 b) {
  return __builtin_shufflevector(a, b, 0,1,2,3,4,5,6,7,8,9,10,11,12,13,14,15);
}

__device__ __forceinline__ bf16x8 ld8(const bf16* p) { return *(const bf16x8*)p; }

__device__ __forceinline__ f32x8 wmma_bf16(bf16x16 a, bf16x16 b, f32x8 c) {
  return __builtin_amdgcn_wmma_f32_16x16x32_bf16(false, a, false, b, (short)0, c, false, false);
}

// ---- CDNA5 async global->LDS (16B per lane), tracked by ASYNCcnt ----
__device__ __forceinline__ void async_g2l(void* lds_ptr, const void* gptr) {
  unsigned loff = (unsigned)(size_t)lds_ptr;   // flat LDS aperture: low 32 bits = LDS offset
  asm volatile("global_load_async_to_lds_b128 %0, %1, off"
               :: "v"(loff), "v"(gptr) : "memory");
}
__device__ __forceinline__ void wait_async() {
  asm volatile("s_wait_asynccnt 0x0" ::: "memory");
}

// ---------------- LayerNorm (fp32 in, bf16 out) ----------------
__global__ __launch_bounds__(256)
void ln_kernel(const float* __restrict__ x, const float* __restrict__ gamma,
               const float* __restrict__ beta, bf16* __restrict__ xn) {
  const int row = blockIdx.x;
  const float* xr = x + (size_t)row * DIMSZ;
  __shared__ float red[256];
  const int tid = threadIdx.x;

  float s = 0.f;
  for (int i = tid; i < DIMSZ; i += 256) s += xr[i];
  red[tid] = s; __syncthreads();
  for (int st = 128; st > 0; st >>= 1) { if (tid < st) red[tid] += red[tid + st]; __syncthreads(); }
  const float mu = red[0] * (1.0f / DIMSZ);
  __syncthreads();

  float v = 0.f;
  for (int i = tid; i < DIMSZ; i += 256) { float d = xr[i] - mu; v += d * d; }
  red[tid] = v; __syncthreads();
  for (int st = 128; st > 0; st >>= 1) { if (tid < st) red[tid] += red[tid + st]; __syncthreads(); }
  const float rstd = rsqrtf(red[0] * (1.0f / DIMSZ) + 1e-5f);

  bf16* xo = xn + (size_t)row * DIMSZ;
  for (int i = tid; i < DIMSZ; i += 256)
    xo[i] = f2bf((xr[i] - mu) * rstd * gamma[i] + beta[i]);
}

// ---------------- fp32 -> bf16 convert ----------------
__global__ __launch_bounds__(256)
void cvt_kernel(const float* __restrict__ src, bf16* __restrict__ dst, int n) {
  for (int i = blockIdx.x * 256 + threadIdx.x; i < n; i += gridDim.x * 256)
    dst[i] = f2bf(src[i]);
}

// ---------------- Tiled GEMM: C[M,N] = A[M,K] * W[N,K]^T ----------------
// Double-buffered async global->LDS staging; 8 waves x (32x32) per 64x128 block tile.
// MODE 0: fp32 output + bias. MODE 1: scatter into q / k / v^T bf16 buffers.
template<int MODE>
__global__ __launch_bounds__(256)
void gemm_bf16(const bf16* __restrict__ A, const bf16* __restrict__ W,
               int M, int Nout, int K,
               float* __restrict__ outF, const float* __restrict__ bias,
               bf16* __restrict__ qout, bf16* __restrict__ kout, bf16* __restrict__ vtout) {
  __shared__ bf16 As[2][64][40];
  __shared__ bf16 Ws[2][128][40];
  const int tid  = threadIdx.x;
  const int wave = tid >> 5, lane = tid & 31;
  const int lo = lane & 15, hi = lane >> 4;
  const int wm = wave & 1, wn = wave >> 1;          // 2 x 4 wave grid, 32x32 per wave
  const int mBase = blockIdx.y * 64;
  const int nBase = blockIdx.x * 128;

  const int arow = tid >> 2, acol = (tid & 3) * 8;  // 64x32 tile, 8 bf16/thread
  const int wrow = tid >> 1, wcol = (tid & 1) * 16; // 128x32 tile, 16 bf16/thread
  const bf16* gA = &A[(size_t)(mBase + arow) * K + acol];
  const bf16* gW = &W[(size_t)(nBase + wrow) * K + wcol];

  f32x8 acc[2][2] = {};

  // prologue: fill buffer 0
  async_g2l(&As[0][arow][acol],   gA);
  async_g2l(&Ws[0][wrow][wcol],   gW);
  async_g2l(&Ws[0][wrow][wcol+8], gW + 8);
  wait_async();
  __syncthreads();

  int buf = 0;
  for (int k0 = 0; k0 < K; k0 += 32) {
    // issue all fragment loads for this K-step first (one ds clause)
    bf16x16 afrag[2], bfrag[2];
#pragma unroll
    for (int i = 0; i < 2; i++) {
      const bf16* ar = &As[buf][wm * 32 + i * 16 + lo][0];
      afrag[i] = join8(ld8(ar + 16 * hi), ld8(ar + 8 + 16 * hi));
    }
#pragma unroll
    for (int j = 0; j < 2; j++) {
      const bf16* br = &Ws[buf][wn * 32 + j * 16 + lo][0];
      bfrag[j] = join8(ld8(br + 16 * hi), ld8(br + 8 + 16 * hi));
    }

    if (k0 + 32 < K) {  // prefetch next K-tile into the other buffer (overlaps WMMAs)
      async_g2l(&As[buf^1][arow][acol],   gA + k0 + 32);
      async_g2l(&Ws[buf^1][wrow][wcol],   gW + k0 + 32);
      async_g2l(&Ws[buf^1][wrow][wcol+8], gW + k0 + 40);
    }

#pragma unroll
    for (int i = 0; i < 2; i++)
#pragma unroll
      for (int j = 0; j < 2; j++)
        acc[i][j] = wmma_bf16(afrag[i], bfrag[j], acc[i][j]);

    wait_async();
    __syncthreads();
    buf ^= 1;
  }

#pragma unroll
  for (int i = 0; i < 2; i++) {
#pragma unroll
    for (int j = 0; j < 2; j++) {
      const int col = nBase + wn * 32 + j * 16 + lo;
      if (MODE == 0) {
        const float bv = bias[col];
#pragma unroll
        for (int r = 0; r < 8; r++) {
          const int row = mBase + wm * 32 + i * 16 + r + 8 * hi;
          outF[(size_t)row * Nout + col] = acc[i][j][r] + bv;
        }
      } else {
        const int which = col >> 10;       // 0:q 1:k 2:v
        const int rem = col & 1023;
        const int h = rem >> 6, d = rem & 63;
#pragma unroll
        for (int r = 0; r < 8; r++) {
          const int row = mBase + wm * 32 + i * 16 + r + 8 * hi;
          const int b = row >> 11, n = row & (NSEQ - 1);
          const float vv = acc[i][j][r];
          const size_t bh = (size_t)(b * NHEAD + h);
          if (which == 0)      qout[(bh * NSEQ + n) * DHEAD + d] = f2bf(vv * 0.125f); // fold 1/sqrt(64)
          else if (which == 1) kout[(bh * NSEQ + n) * DHEAD + d] = f2bf(vv);
          else                 vtout[(bh * DHEAD + d) * NSEQ + n] = f2bf(vv);
        }
      }
    }
  }
}

// ---------------- Flash attention ----------------
// Block = 8 waves sharing one (b,h); each wave owns a 16-query tile.
// K/V tiles for each 32-key step are staged once per block in LDS via
// double-buffered async loads; all 8 waves consume them.
__global__ __launch_bounds__(256)
void attn_kernel(const bf16* __restrict__ q, const bf16* __restrict__ k,
                 const bf16* __restrict__ vt, bf16* __restrict__ attnOut) {
  __shared__ bf16 Ks[2][32][72];   // [key row 0..31][d 0..63]
  __shared__ bf16 Vs[2][64][40];   // [d 0..63][key 0..31]
  __shared__ bf16 plds[8][16][40]; // per-wave P transpose staging
  const int tid  = threadIdx.x;
  const int wave = tid >> 5, lane = tid & 31;
  const int lo = lane & 15, hi = lane >> 4;

  const int bid   = blockIdx.x;                 // 1024 blocks
  const int bh    = bid >> 4;                   // (b*16+h): 0..63
  const int qtile = (bid & 15) * 8 + wave;      // 0..127
  const int m0    = qtile * 16;
  const int b     = bh >> 4, h = bh & 15;

  const bf16* qb  = q  + (size_t)bh * NSEQ * DHEAD;
  const bf16* kb  = k  + (size_t)bh * NSEQ * DHEAD;
  const bf16* vb  = vt + (size_t)bh * DHEAD * NSEQ;

  // cooperative K/V loaders: 8 bf16 (16B) per thread per tile
  const int krow = tid >> 3, kchunk = (tid & 7) * 8;   // K: 32 x 64
  const int vrow = tid >> 2, vchunk = (tid & 3) * 8;   // V: 64 x 32
  const bf16* gK = kb + (size_t)krow * DHEAD + kchunk;
  const bf16* gV = vb + (size_t)vrow * NSEQ + vchunk;

  // Q A-fragments (16 rows x 64 d, as two 16x32 frags); rows = lanes
  bf16x16 aq[2];
  {
    const bf16* qr = qb + (size_t)(m0 + lo) * DHEAD;
#pragma unroll
    for (int i = 0; i < 2; i++)
      aq[i] = join8(ld8(qr + 32 * i + 16 * hi), ld8(qr + 32 * i + 8 + 16 * hi));
  }

  f32x8 o[4] = {};
  float m_r[8], l_r[8];
#pragma unroll
  for (int r = 0; r < 8; r++) { m_r[r] = -1e30f; l_r[r] = 0.f; }

  // prologue: stage keys/values 0..31
  async_g2l(&Ks[0][krow][kchunk], gK);
  async_g2l(&Vs[0][vrow][vchunk], gV);
  wait_async();
  __syncthreads();

  int buf = 0;
  for (int j = 0; j < NSEQ / 32; j++) {
    const int kn = j * 32;
    if (kn + 32 < NSEQ) {  // prefetch next 32 keys
      async_g2l(&Ks[buf^1][krow][kchunk], gK + (size_t)(kn + 32) * DHEAD);
      async_g2l(&Vs[buf^1][vrow][vchunk], gV + (kn + 32));
    }

    // ---- load ALL K B-fragments first (batched ds clause), then 4 WMMAs ----
    bf16x16 bk[2][2];   // [st][tile]
#pragma unroll
    for (int st = 0; st < 2; st++) {
      const bf16* kr0 = &Ks[buf][lo][32 * st + 16 * hi];
      const bf16* kr1 = &Ks[buf][16 + lo][32 * st + 16 * hi];
      bk[st][0] = join8(ld8(kr0), ld8(kr0 + 8));
      bk[st][1] = join8(ld8(kr1), ld8(kr1 + 8));
    }
    f32x8 s0 = {}, s1 = {};
#pragma unroll
    for (int st = 0; st < 2; st++) {
      s0 = wmma_bf16(aq[st], bk[st][0], s0);
      s1 = wmma_bf16(aq[st], bk[st][1], s1);
    }

    // online softmax over the 32 new columns (row = r + 8*hi, spread over 16 lanes)
#pragma unroll
    for (int r = 0; r < 8; r++) {
      float t = fmaxf(s0[r], s1[r]);
      t = fmaxf(t, __shfl_xor(t, 8, 32));
      t = fmaxf(t, __shfl_xor(t, 4, 32));
      t = fmaxf(t, __shfl_xor(t, 2, 32));
      t = fmaxf(t, __shfl_xor(t, 1, 32));
      const float mn = fmaxf(m_r[r], t);
      const float corr = __expf(m_r[r] - mn);
      const float p0 = __expf(s0[r] - mn);
      const float p1 = __expf(s1[r] - mn);
      float rs = p0 + p1;
      rs += __shfl_xor(rs, 8, 32);
      rs += __shfl_xor(rs, 4, 32);
      rs += __shfl_xor(rs, 2, 32);
      rs += __shfl_xor(rs, 1, 32);
      l_r[r] = l_r[r] * corr + rs;
      m_r[r] = mn;
#pragma unroll
      for (int c = 0; c < 4; c++) o[c][r] *= corr;
      s0[r] = p0; s1[r] = p1;
    }

    // C-layout -> A-layout transpose of P through wave-private LDS
#pragma unroll
    for (int r = 0; r < 8; r++) {
      plds[wave][r + 8 * hi][lo]      = f2bf(s0[r]);
      plds[wave][r + 8 * hi][16 + lo] = f2bf(s1[r]);
    }
    const bf16* pr = &plds[wave][lo][0];
    bf16x16 ap = join8(ld8(pr + 16 * hi), ld8(pr + 8 + 16 * hi));

    // ---- load ALL V B-fragments first, then 4 WMMAs ----
    bf16x16 bv[4];
#pragma unroll
    for (int c = 0; c < 4; c++) {
      const bf16* vr = &Vs[buf][16 * c + lo][16 * hi];
      bv[c] = join8(ld8(vr), ld8(vr + 8));
    }
#pragma unroll
    for (int c = 0; c < 4; c++)
      o[c] = wmma_bf16(ap, bv[c], o[c]);

    wait_async();
    __syncthreads();
    buf ^= 1;
  }

  // normalize and store [B, N, H*64] bf16
#pragma unroll
  for (int c = 0; c < 4; c++) {
#pragma unroll
    for (int r = 0; r < 8; r++) {
      const int n = m0 + r + 8 * hi;
      const int col = h * DHEAD + 16 * c + lo;
      attnOut[((size_t)b * NSEQ + n) * (NHEAD * DHEAD) + col] = f2bf(o[c][r] / l_r[r]);
    }
  }
}

// ---------------- host-side launcher ----------------
extern "C" void kernel_launch(void* const* d_in, const int* in_sizes, int n_in,
                              void* d_out, int out_size, void* d_ws, size_t ws_size,
                              hipStream_t stream) {
  const float* x     = (const float*)d_in[0];
  const float* gamma = (const float*)d_in[1];
  const float* beta  = (const float*)d_in[2];
  const float* w_qkv = (const float*)d_in[3];
  const float* w_out = (const float*)d_in[4];
  const float* b_out = (const float*)d_in[5];
  float* out = (float*)d_out;

  char* ws = (char*)d_ws;
  size_t off = 0;
  bf16* xn    = (bf16*)(ws + off); off += (size_t)MROWS * DIMSZ * 2;          // 16 MB
  bf16* wqkvb = (bf16*)(ws + off); off += (size_t)QKVN * DIMSZ * 2;           // 6 MB
  bf16* woutb = (bf16*)(ws + off); off += (size_t)DIMSZ * DIMSZ * 2;          // 2 MB
  bf16* qbuf  = (bf16*)(ws + off); off += (size_t)BATCH * NHEAD * NSEQ * DHEAD * 2; // 16 MB
  bf16* kbuf  = (bf16*)(ws + off); off += (size_t)BATCH * NHEAD * NSEQ * DHEAD * 2; // 16 MB
  bf16* vtbuf = (bf16*)(ws + off); off += (size_t)BATCH * NHEAD * NSEQ * DHEAD * 2; // 16 MB
  bf16* attnb = (bf16*)(ws + off); off += (size_t)MROWS * DIMSZ * 2;          // 16 MB

  ln_kernel<<<MROWS, 256, 0, stream>>>(x, gamma, beta, xn);
  cvt_kernel<<<2048, 256, 0, stream>>>(w_qkv, wqkvb, QKVN * DIMSZ);
  cvt_kernel<<<1024, 256, 0, stream>>>(w_out, woutb, DIMSZ * DIMSZ);

  gemm_bf16<1><<<dim3(QKVN / 128, MROWS / 64), 256, 0, stream>>>(
      xn, wqkvb, MROWS, QKVN, DIMSZ, nullptr, nullptr, qbuf, kbuf, vtbuf);

  attn_kernel<<<BATCH * NHEAD * (NSEQ / 16) / 8, 256, 0, stream>>>(qbuf, kbuf, vtbuf, attnb);

  gemm_bf16<0><<<dim3(DIMSZ / 128, MROWS / 64), 256, 0, stream>>>(
      attnb, woutb, MROWS, DIMSZ, DIMSZ, out, b_out, nullptr, nullptr, nullptr);
}
